// CustomLSTM_50036368998961
// MI455X (gfx1250) — matrix operand order repaired and checked by reading, not compiled, not run on hardware
//
#include <hip/hip_runtime.h>
#include <hip/hip_bf16.h>
#include <cstddef>

// ---------------- types ----------------
typedef __bf16 bf16_t;
typedef bf16_t v16bf __attribute__((ext_vector_type(16)));
typedef float  v8f   __attribute__((ext_vector_type(8)));
typedef unsigned int v4u __attribute__((ext_vector_type(4)));
typedef int v4i_t __attribute__((ext_vector_type(4)));

#define AS1 __attribute__((address_space(1)))
#define AS3 __attribute__((address_space(3)))

#define B_  32
#define S_  1024
#define I_  512
#define H_  512
#define N4_ 2048   // 4*H concatenated gates (i,f,g,o)
#define LDW 520    // LDS row stride in bf16 elems (512 + 8 pad -> bank-conflict-free b128)
#define NBLK_SCAN 8
// dynamic LDS for the scan kernel: U slice (4 gates x 64 rows) + h tile (32 rows)
#define SCAN_LDS_ELEMS (4 * 64 * LDW + 32 * LDW)
#define SCAN_LDS_BYTES (SCAN_LDS_ELEMS * 2)

#if defined(__has_builtin)
#if __has_builtin(__builtin_amdgcn_global_load_async_to_lds_b128) && \
    __has_builtin(__builtin_amdgcn_s_wait_asynccnt)
#define USE_ASYNC_LDS 1
#endif
#endif

// ---------------- helpers ----------------
__device__ __forceinline__ unsigned short f2bf(float f) {
    unsigned int u = __float_as_uint(f);
    u += 0x7FFFu + ((u >> 16) & 1u);        // round-to-nearest-even
    return (unsigned short)(u >> 16);
}
__device__ __forceinline__ float bf2f(unsigned short h) {
    return __uint_as_float(((unsigned int)h) << 16);
}
__device__ __forceinline__ v16bf load_frag(const unsigned short* p0,
                                           const unsigned short* p1) {
    union { v16bf v; v4u u[2]; } f;
    f.u[0] = *reinterpret_cast<const v4u*>(p0);
    f.u[1] = *reinterpret_cast<const v4u*>(p1);
    return f.v;
}
__device__ __forceinline__ float sigm(float x) {
    return 1.0f / (1.0f + __expf(-x));
}
__device__ __forceinline__ void lds_copy16(const unsigned short* src,
                                           unsigned short* dst) {
#if defined(USE_ASYNC_LDS)
    unsigned short* srcnc = const_cast<unsigned short*>(src);
    __builtin_amdgcn_global_load_async_to_lds_b128(
        (AS1 v4i_t*)srcnc, (AS3 v4i_t*)dst, 0, 0);
#else
    *reinterpret_cast<v4u*>(dst) = *reinterpret_cast<const v4u*>(src);
#endif
}
__device__ __forceinline__ void lds_copy_fence() {
#if defined(USE_ASYNC_LDS)
    __builtin_amdgcn_s_wait_asynccnt(0);
#endif
    __syncthreads();
}
// Cooperatively stage `nrows` rows of 512 bf16 (row stride 512) from global
// into LDS with row stride LDW; nthreads participants.
__device__ __forceinline__ void stage_rows(const unsigned short* __restrict__ g,
                                           unsigned short* l, int nrows, int tid) {
    const int nchunks = nrows * 64;                 // 16B chunks
    for (int idx = tid; idx < nchunks; idx += 256) {
        const int row = idx >> 6, col = idx & 63;
        lds_copy16(g + row * 512 + col * 8, l + row * LDW + col * 8);
    }
    lds_copy_fence();
}

// Device-wide barrier for a small resident grid (monotonic counter).
__device__ __forceinline__ void grid_barrier(unsigned int* cnt, unsigned target) {
    __threadfence();                               // release this thread's stores
    __syncthreads();
    if (threadIdx.x == 0) {
        __hip_atomic_fetch_add(cnt, 1u, __ATOMIC_RELEASE, __HIP_MEMORY_SCOPE_AGENT);
        while (__hip_atomic_load(cnt, __ATOMIC_ACQUIRE, __HIP_MEMORY_SCOPE_AGENT) <
               target) {
            __builtin_amdgcn_s_sleep(1);
        }
    }
    __syncthreads();
    (void)__hip_atomic_load(cnt, __ATOMIC_ACQUIRE, __HIP_MEMORY_SCOPE_AGENT);
}

// ---------------- utility kernels ----------------
__global__ void cvt_f32_bf16_kernel(const float* __restrict__ in,
                                    unsigned short* __restrict__ out, int n) {
    int i = blockIdx.x * blockDim.x + threadIdx.x;
    if (i < n) out[i] = f2bf(in[i]);
}
__global__ void copy_f32_kernel(const float* __restrict__ in,
                                float* __restrict__ out, int n) {
    int i = blockIdx.x * blockDim.x + threadIdx.x;
    if (i < n) out[i] = in[i];
}
__global__ void zero_u32_kernel(unsigned int* __restrict__ p, int n) {
    int i = blockIdx.x * blockDim.x + threadIdx.x;
    if (i < n) p[i] = 0u;
}

// ---------------- phase 1: P[s][b][n] = x @ Wcat^T + bcat (bf16 out) ------
__global__ __launch_bounds__(256)
void lstm_xproj_kernel(const unsigned short* __restrict__ xbf,   // (B*S, I) bf16
                       const unsigned short* __restrict__ wbf,   // (2048, 512) bf16
                       const float* __restrict__ bcat,           // (2048) f32
                       unsigned short* __restrict__ P) {         // (S, B, 2048) bf16
    __shared__ unsigned short lx[16 * LDW];
    const int tid  = threadIdx.x;
    const int lane = tid & 31;
    const int wave = tid >> 5;
    const int mtil = blockIdx.x >> 2;           // 0..2047
    const int nblk = blockIdx.x & 3;            // 0..3
    const int m0   = mtil * 16;
    const int n0   = nblk * 512 + wave * 64;
    const int lo16 = lane & 15;
    const int hi   = lane >> 4;
    const int ka   = hi * 16;                   // A-fragment K offset
    const int kb   = hi * 16;                   // B-fragment K offset

    stage_rows(xbf + (size_t)m0 * I_, lx, 16, tid);

    const unsigned short* arow = lx + lo16 * LDW + ka;
    const unsigned short* brow = wbf + (size_t)(n0 + lo16) * I_ + kb;

    v8f acc[4] = {};
    v16bf A0 = load_frag(arow, arow + 8);
    v16bf Bf0[4];
#pragma unroll
    for (int j = 0; j < 4; ++j)
        Bf0[j] = load_frag(brow + j * 16 * I_, brow + j * 16 * I_ + 8);

#pragma unroll
    for (int ks = 0; ks < 16; ++ks) {
        v16bf A1;
        v16bf Bf1[4];
        if (ks < 15) {                           // prefetch next K-step
            const int kn = (ks + 1) * 32;
            A1 = load_frag(arow + kn, arow + kn + 8);
#pragma unroll
            for (int j = 0; j < 4; ++j)
                Bf1[j] = load_frag(brow + j * 16 * I_ + kn,
                                   brow + j * 16 * I_ + kn + 8);
        }
#pragma unroll
        for (int j = 0; j < 4; ++j)
            acc[j] = __builtin_amdgcn_wmma_f32_16x16x32_bf16(
                false, A0, false, Bf0[j], (short)0, acc[j], false, false);
        if (ks < 15) {
            A0 = A1;
#pragma unroll
            for (int j = 0; j < 4; ++j) Bf0[j] = Bf1[j];
        }
    }

#pragma unroll
    for (int j = 0; j < 4; ++j) {
        const int n  = n0 + j * 16 + lo16;
        const float bv = bcat[n];
#pragma unroll
        for (int e = 0; e < 8; ++e) {
            const int mabs = m0 + e + 8 * hi;   // row in (B*S)
            const int b = mabs >> 10;           // / S_
            const int s = mabs & 1023;          // % S_
            P[((size_t)s * B_ + b) * N4_ + n] = f2bf(acc[j][e] + bv);
        }
    }
}

// ---------------- phase 2: persistent recurrent scan ------------------------
// 8 resident blocks (one per WGP), 64 waves. Each block keeps its 64-column
// slice of all four U gate matrices RESIDENT in LDS (256KB) for the whole
// 1024-step scan; per step, h (32x512 bf16) is async-staged to LDS and the
// K-loop is pure ds_load_b128 -> v_wmma. One device-wide barrier per step;
// h double-buffered in global.
__global__ __launch_bounds__(256)
void lstm_scan_kernel(const unsigned short* __restrict__ P,    // (S, B, 2048) bf16
                      const unsigned short* __restrict__ ubf,  // (2048, 512) bf16
                      unsigned short* __restrict__ hb0,        // (B, H) bf16
                      unsigned short* __restrict__ hb1,        // (B, H) bf16
                      float* __restrict__ c,                   // (B, H) f32
                      float* __restrict__ out_hs,              // (B, S, H) f32
                      unsigned int* __restrict__ bar) {
    extern __shared__ unsigned short smem[];
    unsigned short* lu  = smem;                   // [4*64][LDW] resident U slice
    unsigned short* lha = smem + 4 * 64 * LDW;    // [32][LDW] staged h

    const int tid  = threadIdx.x;
    const int lane = tid & 31;
    const int wave = tid >> 5;
    const int mt   = wave & 1;                    // batch half
    const int ntl  = wave >> 1;                   // 0..3 local n-tile
    const int nb0  = blockIdx.x * 64;             // block's first h column
    const int m0   = mt * 16;
    const int n0   = nb0 + ntl * 16;              // global h column base
    const int lo16 = lane & 15;
    const int hi   = lane >> 4;
    const int ka   = hi * 16;
    const int kb   = hi * 16;

    // ---- one-time stage of the block's U slice: 4 gates x 64 rows x 512 ----
    {
        const int nchunks = 4 * 64 * 64;          // rows x 16B-chunks
        for (int idx = tid; idx < nchunks; idx += 256) {
            const int row = idx >> 6;             // 0..255: g*64 + r
            const int col = idx & 63;
            const int g = row >> 6, r = row & 63;
            lds_copy16(ubf + (size_t)(g * H_ + nb0 + r) * H_ + col * 8,
                       lu + row * LDW + col * 8);
        }
        lds_copy_fence();
    }

    const unsigned short* arow = lha + (m0 + lo16) * LDW + ka;
    const unsigned short* brow = lu + (ntl * 16 + lo16) * LDW + kb;
    const int n = n0 + lo16;

    for (int t = 0; t < S_; ++t) {
        grid_barrier(bar, (unsigned)(NBLK_SCAN * (t + 1)));
        const unsigned short* hin  = (t & 1) ? hb1 : hb0;
        unsigned short*       hout = (t & 1) ? hb0 : hb1;
        stage_rows(hin, lha, 32, tid);

        v8f acc[4] = {};
        v16bf A0 = load_frag(arow, arow + 8);
        v16bf Bf0[4];
#pragma unroll
        for (int g = 0; g < 4; ++g)
            Bf0[g] = load_frag(brow + g * 64 * LDW, brow + g * 64 * LDW + 8);

#pragma unroll
        for (int ks = 0; ks < 16; ++ks) {
            v16bf A1;
            v16bf Bf1[4];
            if (ks < 15) {                        // prefetch next K-step (LDS)
                const int kn = (ks + 1) * 32;
                A1 = load_frag(arow + kn, arow + kn + 8);
#pragma unroll
                for (int g = 0; g < 4; ++g)
                    Bf1[g] = load_frag(brow + g * 64 * LDW + kn,
                                       brow + g * 64 * LDW + kn + 8);
            }
#pragma unroll
            for (int g = 0; g < 4; ++g)
                acc[g] = __builtin_amdgcn_wmma_f32_16x16x32_bf16(
                    false, A0, false, Bf0[g], (short)0, acc[g], false, false);
            if (ks < 15) {
                A0 = A1;
#pragma unroll
                for (int g = 0; g < 4; ++g) Bf0[g] = Bf1[g];
            }
        }

        // fused gates + cell update (D layout: m = e + 8*hi, n fixed per lane)
        const unsigned short* Pt = P + (size_t)t * B_ * N4_;
#pragma unroll
        for (int e = 0; e < 8; ++e) {
            const int b = m0 + e + 8 * hi;        // 0..31
            const size_t pb = (size_t)b * N4_ + n;
            float pi = bf2f(Pt[pb + 0 * H_]) + acc[0][e];
            float pf = bf2f(Pt[pb + 1 * H_]) + acc[1][e];
            float pg = bf2f(Pt[pb + 2 * H_]) + acc[2][e];
            float po = bf2f(Pt[pb + 3 * H_]) + acc[3][e];
            float it = sigm(pi);
            float ft = sigm(pf);
            float gt = tanhf(pg);
            float ot = sigm(po);
            const size_t ci = (size_t)b * H_ + n;
            float cn = ft * c[ci] + it * gt;
            c[ci] = cn;
            float hn = ot * tanhf(cn);
            out_hs[((size_t)b * S_ + t) * H_ + n] = hn;
            hout[ci] = f2bf(hn);
        }
    }
}

// ---------------- tail: emit (h_T, c_T) ------------------------------------
__global__ void lstm_tail_kernel(const float* __restrict__ out_hs,
                                 const float* __restrict__ c,
                                 float* __restrict__ out) {
    int i = blockIdx.x * blockDim.x + threadIdx.x;   // 0..16383
    if (i < B_ * H_) {
        int b = i >> 9, n = i & 511;
        out[(size_t)B_ * S_ * H_ + i] =
            out_hs[((size_t)b * S_ + (S_ - 1)) * H_ + n];
        out[(size_t)B_ * S_ * H_ + B_ * H_ + i] = c[i];
    }
}

// ---------------- launch ----------------------------------------------------
extern "C" void kernel_launch(void* const* d_in, const int* in_sizes, int n_in,
                              void* d_out, int out_size, void* d_ws, size_t ws_size,
                              hipStream_t stream) {
    (void)in_sizes; (void)n_in; (void)out_size; (void)ws_size;
    const float* x  = (const float*)d_in[0];
    const float* Wg[4] = { (const float*)d_in[1], (const float*)d_in[4],
                           (const float*)d_in[7], (const float*)d_in[10] };
    const float* bg[4] = { (const float*)d_in[2], (const float*)d_in[5],
                           (const float*)d_in[8], (const float*)d_in[11] };
    const float* Ug[4] = { (const float*)d_in[3], (const float*)d_in[6],
                           (const float*)d_in[9], (const float*)d_in[12] };
    float* out = (float*)d_out;

    // workspace layout (bytes)
    char* ws = (char*)d_ws;
    const size_t SZ_XBF = (size_t)B_ * S_ * I_ * 2;        // 32 MB
    const size_t SZ_W   = (size_t)N4_ * I_ * 2;            // 2 MB
    const size_t SZ_U   = (size_t)N4_ * H_ * 2;            // 2 MB
    const size_t SZ_B   = (size_t)N4_ * 4;                 // 8 KB
    const size_t SZ_C   = (size_t)B_ * H_ * 4;             // 64 KB
    const size_t SZ_H   = (size_t)B_ * H_ * 2;             // 32 KB
    const size_t SZ_BAR = 256;
    unsigned short* xbf = (unsigned short*)(ws);
    unsigned short* wbf = (unsigned short*)(ws + SZ_XBF);
    unsigned short* ubf = (unsigned short*)(ws + SZ_XBF + SZ_W);
    float*          bc  = (float*)        (ws + SZ_XBF + SZ_W + SZ_U);
    float*          cst = (float*)        (ws + SZ_XBF + SZ_W + SZ_U + SZ_B);
    unsigned short* hb0 = (unsigned short*)((char*)cst + SZ_C);
    unsigned short* hb1 = (unsigned short*)((char*)hb0 + SZ_H);
    unsigned int*   bar = (unsigned int*)  ((char*)hb1 + SZ_H);
    unsigned short* P   = (unsigned short*)((char*)bar + SZ_BAR); // 128 MB

    // --- precision conversion: x, W, U -> bf16; biases -> concat fp32 ---
    {
        int n = B_ * S_ * I_;
        cvt_f32_bf16_kernel<<<(n + 255) / 256, 256, 0, stream>>>(x, xbf, n);
    }
    for (int g = 0; g < 4; ++g) {
        int n = H_ * I_;
        cvt_f32_bf16_kernel<<<(n + 255) / 256, 256, 0, stream>>>(
            Wg[g], wbf + (size_t)g * H_ * I_, n);
        cvt_f32_bf16_kernel<<<(n + 255) / 256, 256, 0, stream>>>(
            Ug[g], ubf + (size_t)g * H_ * H_, n);
        copy_f32_kernel<<<(H_ + 255) / 256, 256, 0, stream>>>(
            bg[g], bc + g * H_, H_);
    }
    // zero c, h ping-pong buffers, and barrier counter (contiguous region)
    {
        int n = (int)((SZ_C + 2 * SZ_H + SZ_BAR) / 4);
        zero_u32_kernel<<<(n + 255) / 256, 256, 0, stream>>>(
            (unsigned int*)cst, n);
    }

    // --- phase 1: all-timestep input projections (parallel bf16 WMMA GEMM) ---
    lstm_xproj_kernel<<<8192, 256, 0, stream>>>(xbf, wbf, bc, P);

    // --- phase 2: persistent scan kernel, U resident in 300KB LDS/WGP ---
    hipFuncSetAttribute((const void*)lstm_scan_kernel,
                        hipFuncAttributeMaxDynamicSharedMemorySize,
                        SCAN_LDS_BYTES);
    lstm_scan_kernel<<<NBLK_SCAN, 256, SCAN_LDS_BYTES, stream>>>(
        P, ubf, hb0, hb1, cst, out, bar);

    // --- tail: h_T (from hidden_seq) and c_T ---
    lstm_tail_kernel<<<(B_ * H_ + 255) / 256, 256, 0, stream>>>(out, cst, out);
}
